// EMASplitDelta_35588099014928
// MI455X (gfx1250) — matrix operand order repaired and checked by reading, not compiled, not run on hardware
//
#include <hip/hip_runtime.h>
#include <hip/hip_bf16.h>
#include <stdint.h>

#define B_    8
#define L_    2048
#define H_    512
#define V_    32000
#define HALF_ 256

typedef __attribute__((ext_vector_type(16))) __bf16 v16bf;
typedef __attribute__((ext_vector_type(8)))  __bf16 bf16x8;
typedef __attribute__((ext_vector_type(8)))  float  v8f;

// ---- CDNA5 async global->LDS copy (ASYNCcnt path), guarded by __has_builtin
#if defined(__has_builtin)
#if __has_builtin(__builtin_amdgcn_global_load_async_to_lds_b32) && \
    __has_builtin(__builtin_amdgcn_s_wait_asynccnt)
#define USE_ASYNC_LDS 1
#endif
#endif
#ifndef USE_ASYNC_LDS
#define USE_ASYNC_LDS 0
#endif

#if USE_ASYNC_LDS
typedef __attribute__((address_space(1))) int g_i32;   // global
typedef __attribute__((address_space(3))) int l_i32;   // LDS
__device__ __forceinline__ void async_copy_b32(const float* g, float* l) {
  __builtin_amdgcn_global_load_async_to_lds_b32((g_i32*)g, (l_i32*)l, 0, 0);
}
#endif

// ---------------- WMMA fragment loaders (wave32; A/B use contiguous k per lane half) ----------------
// A (16x32 bf16, MxK): lanes 0-15 -> M=lane, K=[k0..k0+15];
//                      lanes 16-31 -> M=lane-16, K=[k0+16..k0+31].
__device__ __forceinline__ v16bf load_a_frag(const __bf16* __restrict__ A, int lda,
                                             int row0, int k0, int lane) {
  int m  = row0 + (lane & 15);
  int hi = lane >> 4;
  const __bf16* p = A + (size_t)m * lda + k0 + hi * 16;
  union { v16bf v; bf16x8 h[2]; } u;
  u.h[0] = *(const bf16x8*)(p);
  u.h[1] = *(const bf16x8*)(p + 8);
  return u.v;
}

// B (32x16 bf16, KxN) with W stored [N,K] row-major (W^T): lane n = col,
// lanes 0-15 -> K=[k0..k0+15], lanes 16-31 -> K=[k0+16..k0+31]: one 32B load.
__device__ __forceinline__ v16bf load_b_frag(const __bf16* __restrict__ W, int ldw,
                                             int col0, int k0, int lane) {
  int n  = col0 + (lane & 15);
  int hi = lane >> 4;
  return *(const v16bf*)(W + (size_t)n * ldw + k0 + hi * 16);
}

// C/D f32 16x16: VGPR r -> M = r + (lane<16 ? 0 : 8), N = lane&15
// Branch-free epilogue: bias hoisted (row-invariant), address strength-reduced.
template<bool RELU_BF16, bool RESID, bool HAS_BIAS>
__device__ __forceinline__ void store_tile(v8f acc, int m0, int n0,
    const float* __restrict__ bias, const float* __restrict__ resid,
    float* __restrict__ outF, __bf16* __restrict__ outB, int N, int lane) {
  const int n  = n0 + (lane & 15);
  const int mb = m0 + ((lane >> 4) * 8);
  const float bv = HAS_BIAS ? bias[n] : 0.0f;
  size_t idx = (size_t)mb * N + n;
#pragma unroll
  for (int r = 0; r < 8; ++r) {
    float v = acc[r] + bv;
    if (RESID) v += resid[idx];
    if (RELU_BF16) outB[idx] = (__bf16)fmaxf(v, 0.0f);
    else           outF[idx] = v;
    idx += (size_t)N;
  }
}

// ---------------- tiled WMMA GEMM: out = epilogue(A[M,K] x W[N,K]^T) --------
// 8 waves/WG; wave tile 32x64 (8 wmma accs -> 8 loads : 8 wmma per K-step);
// WG tile 128(M) x 128(N).
template<bool RELU_BF16, bool RESID, bool HAS_BIAS>
__global__ __launch_bounds__(256) void wmma_gemm_kernel(
    const __bf16* __restrict__ A, const __bf16* __restrict__ W,
    const float* __restrict__ bias, const float* __restrict__ resid,
    float* __restrict__ outF, __bf16* __restrict__ outB,
    int M, int N, int K)
{
  const int tid  = threadIdx.x;
  const int lane = tid & 31;
  const int wid  = tid >> 5;
  const int wm = wid & 3, wn = wid >> 2;
  const int m_base = blockIdx.y * 128 + wm * 32;
  const int n_base = blockIdx.x * 128 + wn * 64;
  (void)M;

  v8f acc[2][4] = {};
#pragma unroll 2
  for (int k0 = 0; k0 < K; k0 += 32) {
    v16bf a[2];
    a[0] = load_a_frag(A, K, m_base,      k0, lane);
    a[1] = load_a_frag(A, K, m_base + 16, k0, lane);
    v16bf bf[4];
#pragma unroll
    for (int j = 0; j < 4; ++j) bf[j] = load_b_frag(W, K, n_base + 16 * j, k0, lane);
#pragma unroll
    for (int i = 0; i < 2; ++i)
#pragma unroll
      for (int j = 0; j < 4; ++j)
        acc[i][j] = __builtin_amdgcn_wmma_f32_16x16x32_bf16(
            false, a[i], false, bf[j], (short)0, acc[i][j], false, false);
  }
#pragma unroll
  for (int i = 0; i < 2; ++i)
#pragma unroll
    for (int j = 0; j < 4; ++j)
      store_tile<RELU_BF16, RESID, HAS_BIAS>(acc[i][j], m_base + 16 * i,
                                             n_base + 16 * j, bias, resid,
                                             outF, outB, N, lane);
}

// ---------------- helper kernels -------------------------------------------
__global__ void f32_to_bf16_kernel(const float* __restrict__ src,
                                   __bf16* __restrict__ dst, int n) {
  int i = blockIdx.x * blockDim.x + threadIdx.x;
  if (i < n) dst[i] = (__bf16)src[i];
}

__global__ __launch_bounds__(256) void gather_kernel(const int* __restrict__ seq,
    const float* __restrict__ embed, float* __restrict__ hf, __bf16* __restrict__ hb) {
  int token = blockIdx.x, tid = threadIdx.x;
  int v = seq[token];
  const float* src = embed + (size_t)v * H_;
  float*  dst = hf + (size_t)token * H_;
  __bf16* db  = hb + (size_t)token * H_;
  for (int i = tid; i < H_; i += 256) {
    float x = src[i];
    dst[i] = x;
    db[i]  = (__bf16)x;
  }
}

__global__ __launch_bounds__(256) void ln_kernel(const float* __restrict__ x,
    const float* __restrict__ gamma, const float* __restrict__ beta,
    __bf16* __restrict__ hln) {
  int row = blockIdx.x, tid = threadIdx.x;
  const float* xr = x + (size_t)row * H_;
  float a0 = xr[tid], a1 = xr[tid + 256];
  float s1 = a0 + a1, s2 = a0 * a0 + a1 * a1;
#pragma unroll
  for (int m = 16; m > 0; m >>= 1) {         // wave32 butterfly
    s1 += __shfl_xor(s1, m);
    s2 += __shfl_xor(s2, m);
  }
  __shared__ float p1[8], p2[8];
  if ((tid & 31) == 0) { p1[tid >> 5] = s1; p2[tid >> 5] = s2; }
  __syncthreads();
  float t1 = 0.f, t2 = 0.f;
#pragma unroll
  for (int w = 0; w < 8; ++w) { t1 += p1[w]; t2 += p2[w]; }
  float mu  = t1 * (1.0f / H_);
  float var = t2 * (1.0f / H_) - mu * mu;
  float inv = rsqrtf(var + 1e-5f);
  __bf16* o = hln + (size_t)row * H_;
  o[tid]       = (__bf16)((a0 - mu) * inv * gamma[tid]       + beta[tid]);
  o[tid + 256] = (__bf16)((a1 - mu) * inv * gamma[tid + 256] + beta[tid + 256]);
}

// ---------------- delta-rule EMA scan --------------------------------------
// 16 blocks = (batch, {sem,epi}); state M[256][256] fp32 padded to stride 257
// stays resident in the 320KB WGP LDS for all 2047 serial steps.
// k_{t+1} is prefetched into a double-buffered LDS slot with
// GLOBAL_LOAD_ASYNC_TO_LDS_B32 while step t computes; the s_wait_asynccnt is
// folded into the end-of-step barrier, hiding the per-step global latency.
__global__ __launch_bounds__(256) void scan_kernel(const float* __restrict__ k_all,
                                                   float* __restrict__ c) {
  extern __shared__ float smem[];
  float* Mm   = smem;                     // 256*257
  float* kbuf = smem + 256 * 257;         // 2 x 256 double-buffered k_t
  float* kn   = kbuf + 512;               // 256
  float* red  = kn + 256;                 // 8 wave partials
  const int tid   = threadIdx.x;
  const int lane  = tid & 31;
  const int wv    = tid >> 5;
  const int b     = blockIdx.x >> 1;
  const int which = blockIdx.x & 1;
  const float* kb = k_all + (size_t)b * L_ * H_ + which * HALF_;

  for (int i = tid; i < 256 * 257; i += 256) Mm[i] = 0.0f;
#if USE_ASYNC_LDS
  async_copy_b32(kb + tid, kbuf + tid);           // k_0 -> buf0
  __builtin_amdgcn_s_wait_asynccnt(0);
#else
  kbuf[tid] = kb[tid];
#endif
  __syncthreads();

  float* row = Mm + tid * 257;
  for (int t = 0; t < L_ - 1; ++t) {
    const int cur = (t & 1) * 256;
    const int nxt = 256 - cur;
    float kx = kbuf[cur + tid];
#if USE_ASYNC_LDS
    async_copy_b32(kb + (size_t)(t + 1) * H_ + tid, kbuf + nxt + tid);
#else
    float kx_next = kb[(size_t)(t + 1) * H_ + tid];   // issued early, used at bottom
#endif
    // ||k||^2 : wave32 butterfly + one barrier
    float ss = kx * kx;
#pragma unroll
    for (int m = 16; m > 0; m >>= 1) ss += __shfl_xor(ss, m);
    if (lane == 0) red[wv] = ss;
    __syncthreads();
    float tot = 0.f;
#pragma unroll
    for (int w = 0; w < 8; ++w) tot += red[w];
    float inv = 1.0f / fmaxf(sqrtf(tot), 1e-12f);
    kn[tid] = kx * inv;
    __syncthreads();
    // v = M[row] . kn ; M[row] += (1-alpha)*(k - v) * kn
    float v = 0.0f;
#pragma unroll 8
    for (int j = 0; j < 256; ++j) v = fmaf(row[j], kn[j], v);
    float coef = 0.05f * (kx - v);
#pragma unroll 8
    for (int j = 0; j < 256; ++j) row[j] = fmaf(coef, kn[j], row[j]);
#if USE_ASYNC_LDS
    __builtin_amdgcn_s_wait_asynccnt(0);            // own wave's prefetch done
#else
    kbuf[nxt + tid] = kx_next;
#endif
    __syncthreads();
  }
  // final readout uses the RAW (un-normalized) last key, already in kbuf
  const int p = ((L_ - 1) & 1) * 256;
  float v = 0.0f;
#pragma unroll 8
  for (int j = 0; j < 256; ++j) v = fmaf(row[j], kbuf[p + j], v);
  c[b * H_ + which * HALF_ + tid] = v;
}

// ---------------- head: r = c@Wrp^T + brp ; out = r@Wout^T + bout ----------
__global__ __launch_bounds__(256) void rproj_kernel(const float* __restrict__ c,
    const float* __restrict__ Wrp, const float* __restrict__ brp,
    float* __restrict__ r) {
  int idx = blockIdx.x * 256 + threadIdx.x;   // 4096 = B*H
  int b = idx >> 9, n = idx & 511;
  const float* cb = c + b * H_;
  const float* wr = Wrp + (size_t)n * H_;
  float acc = brp[n];
  for (int j = 0; j < H_; ++j) acc = fmaf(cb[j], wr[j], acc);
  r[idx] = acc;
}

__global__ __launch_bounds__(256) void outproj_kernel(const float* __restrict__ r,
    const float* __restrict__ Wout, const float* __restrict__ bout,
    float* __restrict__ out) {
  __shared__ float rs[B_ * H_];               // 16 KB
  int tid = threadIdx.x;
  for (int i = tid; i < B_ * H_; i += 256) rs[i] = r[i];
  __syncthreads();
  int n = blockIdx.x * 256 + tid;             // 32000 cols, Wout read exactly once
  const float* w = Wout + (size_t)n * H_;
  float acc[B_];
#pragma unroll
  for (int b = 0; b < B_; ++b) acc[b] = 0.0f;
  for (int j = 0; j < H_; ++j) {
    float wv = w[j];
#pragma unroll
    for (int b = 0; b < B_; ++b) acc[b] = fmaf(rs[b * H_ + j], wv, acc[b]);
  }
  float bv = bout[n];
#pragma unroll
  for (int b = 0; b < B_; ++b) out[(size_t)b * V_ + n] = acc[b] + bv;
}

// ---------------- launch ----------------------------------------------------
extern "C" void kernel_launch(void* const* d_in, const int* in_sizes, int n_in,
                              void* d_out, int out_size, void* d_ws, size_t ws_size,
                              hipStream_t stream) {
  (void)in_sizes; (void)n_in; (void)out_size; (void)ws_size;
  const int*   seq   = (const int*)  d_in[0];
  const float* embed = (const float*)d_in[1];
  const float* W1    = (const float*)d_in[2];
  const float* b1    = (const float*)d_in[3];
  const float* W2    = (const float*)d_in[4];
  const float* b2    = (const float*)d_in[5];
  const float* gamma = (const float*)d_in[6];
  const float* beta  = (const float*)d_in[7];
  const float* Ws    = (const float*)d_in[8];
  const float* We    = (const float*)d_in[9];
  const float* Wrp   = (const float*)d_in[10];
  const float* brp   = (const float*)d_in[11];
  const float* Wout  = (const float*)d_in[12];
  const float* bout  = (const float*)d_in[13];
  float* out = (float*)d_out;

  char* ws = (char*)d_ws;
  size_t off = 0;
  auto alloc = [&](size_t bytes) -> char* {
    char* p = ws + off;
    off += (bytes + 255) & ~(size_t)255;
    return p;
  };
  const int NT = B_ * L_;                                    // 16384 tokens
  float*  h_f32 = (float*) alloc((size_t)NT * H_ * 4);       // gathered embeds (residual)
  __bf16* hb    = (__bf16*)alloc((size_t)NT * H_ * 2);       // bf16 copy for WMMA
  __bf16* act1  = (__bf16*)alloc((size_t)NT * 2 * H_ * 2);   // relu(h W1^T)
  float*  xres  = (float*) alloc((size_t)NT * H_ * 4);       // h + ffn
  __bf16* hln   = (__bf16*)alloc((size_t)NT * H_ * 2);       // layernormed
  float*  k_all = (float*) alloc((size_t)NT * H_ * 4);       // [ks|ke] per token
  float*  cbuf  = (float*) alloc((size_t)B_ * H_ * 4);
  float*  rbuf  = (float*) alloc((size_t)B_ * H_ * 4);
  __bf16* w1b   = (__bf16*)alloc((size_t)2 * H_ * H_ * 2);
  __bf16* w2b   = (__bf16*)alloc((size_t)2 * H_ * H_ * 2);
  __bf16* wseb  = (__bf16*)alloc((size_t)H_ * H_ * 2);       // [Ws; We] rows

  // weight conversions (one-time, tiny)
  f32_to_bf16_kernel<<<(2 * H_ * H_ + 255) / 256, 256, 0, stream>>>(W1, w1b, 2 * H_ * H_);
  f32_to_bf16_kernel<<<(2 * H_ * H_ + 255) / 256, 256, 0, stream>>>(W2, w2b, 2 * H_ * H_);
  f32_to_bf16_kernel<<<(HALF_ * H_ + 255) / 256, 256, 0, stream>>>(Ws, wseb, HALF_ * H_);
  f32_to_bf16_kernel<<<(HALF_ * H_ + 255) / 256, 256, 0, stream>>>(We, wseb + HALF_ * H_, HALF_ * H_);

  gather_kernel<<<NT, 256, 0, stream>>>(seq, embed, h_f32, hb);

  // FFN GEMM1: [NT,512] x [1024,512]^T + b1 -> relu -> bf16
  wmma_gemm_kernel<true, false, true><<<dim3((2 * H_) / 128, NT / 128), 256, 0, stream>>>(
      hb, w1b, b1, nullptr, nullptr, act1, NT, 2 * H_, H_);
  // FFN GEMM2: [NT,1024] x [512,1024]^T + b2 + residual -> f32
  wmma_gemm_kernel<false, true, true><<<dim3(H_ / 128, NT / 128), 256, 0, stream>>>(
      act1, w2b, b2, h_f32, xres, nullptr, NT, H_, 2 * H_);

  ln_kernel<<<NT, 256, 0, stream>>>(xres, gamma, beta, hln);

  // k projections (Ws & We fused as one [512,512] weight, no bias)
  wmma_gemm_kernel<false, false, false><<<dim3(H_ / 128, NT / 128), 256, 0, stream>>>(
      hln, wseb, nullptr, nullptr, k_all, nullptr, NT, H_, H_);

  // 16 independent recurrences, 256KB state each resident in WGP LDS
  size_t scan_lds = (size_t)(256 * 257 + 2 * 256 + 256 + 64) * sizeof(float);
  scan_kernel<<<16, 256, scan_lds, stream>>>(k_all, cbuf);

  rproj_kernel<<<(B_ * H_) / 256, 256, 0, stream>>>(cbuf, Wrp, brp, rbuf);
  outproj_kernel<<<V_ / 256, 256, 0, stream>>>(rbuf, Wout, bout, out);
}
